// MultiScaleSetAbstractionModule_2173253452297
// MI455X (gfx1250) — matrix operand order repaired and checked by reading, not compiled, not run on hardware
//
#include <hip/hip_runtime.h>
#include <hip/hip_bf16.h>

#define N_ 16384
#define M_ 2048
#define B_ 4
#define C_ 16

typedef __attribute__((ext_vector_type(16))) _Float16 v16h;
typedef __attribute__((ext_vector_type(8)))  _Float16 v8h;
typedef __attribute__((ext_vector_type(8)))  float    v8f;

// Prepped layer: f16 weights padded to KD-multiple-of-32, BN folded to affine.
struct LayerPP {
    const _Float16* Wh;   // [d_out][KD] f16, zero padded
    const float*    sc;   // gamma * rsqrt(var+eps)
    const float*    bs;   // beta - mean * sc
};

// ---------------------------------------------------------------------------
// Kernel 0: per-layer weight prep. f32 W[dout][din] -> f16 Wh[dout][dp] (zero
// pad), and fold BN into per-channel affine. Runs once, tiny.
// ---------------------------------------------------------------------------
__global__ void prep_layer(const float* __restrict__ W,
                           const float* __restrict__ gamma,
                           const float* __restrict__ beta,
                           const float* __restrict__ mean,
                           const float* __restrict__ var,
                           int din, int dout, int dp,
                           _Float16* __restrict__ Wh,
                           float* __restrict__ sc, float* __restrict__ bs)
{
    int i = blockIdx.x * blockDim.x + threadIdx.x;
    int total = dout * dp;
    if (i < total) {
        int o = i / dp, c = i % dp;
        Wh[i] = (c < din) ? (_Float16)W[o * din + c] : (_Float16)0.0f;
    }
    if (i < dout) {
        float s = gamma[i] * rsqrtf(var[i] + 1e-5f);
        sc[i] = s;
        bs[i] = beta[i] - mean[i] * s;
    }
}

// ---------------------------------------------------------------------------
// Kernel 1: farthest point sampling (sequential, latency-bound). One
// 1024-thread block per batch, coords cached in registers, LDS argmax tree.
// ---------------------------------------------------------------------------
__global__ __launch_bounds__(1024) void fps_kernel(const float* __restrict__ pts,
                                                   float* __restrict__ qpts,
                                                   float* __restrict__ out)
{
    __shared__ float sd[1024];
    __shared__ int   si[1024];
    const int b   = blockIdx.x;
    const int tid = threadIdx.x;
    const float* px = pts + (size_t)b * 3 * N_;
    const float* py = px + N_;
    const float* pz = py + N_;

    float x[16], y[16], z[16], dl[16];
#pragma unroll
    for (int i = 0; i < 16; ++i) {
        int n = tid + i * 1024;
        x[i] = px[n]; y[i] = py[n]; z[i] = pz[n];
        dl[i] = 3.4e38f;
    }

    int last = 0;
    for (int m = 0; m < M_; ++m) {
        float lx = px[last], ly = py[last], lz = pz[last];
        if (tid == 0) {
            out[(size_t)(b * 3 + 0) * M_ + m] = lx;
            out[(size_t)(b * 3 + 1) * M_ + m] = ly;
            out[(size_t)(b * 3 + 2) * M_ + m] = lz;
            float* qp = qpts + (size_t)(b * M_ + m) * 3;
            qp[0] = lx; qp[1] = ly; qp[2] = lz;
        }
        float best = -1.0f; int bi = tid;
#pragma unroll
        for (int i = 0; i < 16; ++i) {
            float dx = x[i] - lx, dy = y[i] - ly, dz = z[i] - lz;
            float d2 = dx * dx + dy * dy + dz * dz;
            dl[i] = fminf(dl[i], d2);
            if (dl[i] > best) { best = dl[i]; bi = tid + i * 1024; }
        }
        sd[tid] = best; si[tid] = bi;
        __syncthreads();
        for (int s = 512; s > 0; s >>= 1) {
            if (tid < s && sd[tid + s] > sd[tid]) {
                sd[tid] = sd[tid + s]; si[tid] = si[tid + s];
            }
            __syncthreads();
        }
        last = si[0];
        __syncthreads();
    }
}

// ---------------------------------------------------------------------------
// WMMA fragment loaders. Per the ISA A/B layouts each lane's 16 halves are
// two contiguous 8-half (16B) chunks -> 2 wide loads per fragment.
// ---------------------------------------------------------------------------
__device__ inline v16h cat8(v8h lo, v8h hi)
{
    return __builtin_shufflevector(lo, hi, 0, 1, 2, 3, 4, 5, 6, 7,
                                           8, 9, 10, 11, 12, 13, 14, 15);
}

// A 16x32 f16: lane<16 holds K {0..7,16..23}, lane>=16 holds K {8..15,24..31}.
__device__ inline v16h load_a_frag(const _Float16* __restrict__ src, int ld,
                                   int k0, int i0, int lane)
{
    const int m = lane & 15;
    const int hbase = (lane >= 16) ? 8 : 0;
    const v8h* p = (const v8h*)(src + (k0 + m) * ld + i0 + hbase);
    return cat8(p[0], p[2]);          // +0 and +16 halves
}

// B 32x16 f16: lane<16 -> K {0..7,16..23}, N=lane; lane>=16 -> K {8..15,24..31}, N=lane-16.
// Wh rows are contiguous padded d_in -> two 8-half runs per lane.
__device__ inline v16h load_b_frag(const _Float16* __restrict__ Wh, int ld,
                                   int o0, int i0, int lane)
{
    const int n  = lane & 15;
    const int kb = i0 + ((lane >= 16) ? 8 : 0);
    const v8h* p = (const v8h*)(Wh + (o0 + n) * ld + kb);
    return cat8(p[0], p[2]);
}

// One MLP layer: Y[k][o] = relu(sc[o]*(X·W^T)[k][o] + bs[o]), f16 out to LDS.
// KD = padded reduce dim (src row stride == Wh row stride).
template<int K, int KD, int DOUT>
__device__ inline void run_layer(const _Float16* __restrict__ src, const LayerPP L,
                                 _Float16* __restrict__ dst, int lane)
{
    const int  m  = lane & 15;
    const bool hi = lane >= 16;
    for (int k0 = 0; k0 < K; k0 += 16) {
        for (int o0 = 0; o0 < DOUT; o0 += 16) {
            v8f acc = {};
#pragma unroll
            for (int i0 = 0; i0 < KD; i0 += 32) {
                v16h a  = load_a_frag(src, KD, k0, i0, lane);
                v16h bf = load_b_frag(L.Wh, KD, o0, i0, lane);
                acc = __builtin_amdgcn_wmma_f32_16x16x32_f16(
                    false, a, false, bf, (short)0, acc, false, false);
            }
            const int o = o0 + m;
            const float sc = L.sc[o];
            const float bs = L.bs[o];
            const int rbase = k0 + (hi ? 8 : 0);
#pragma unroll
            for (int j = 0; j < 8; ++j) {
                float y = fmaxf(acc[j] * sc + bs, 0.0f);
                dst[(rbase + j) * DOUT + o] = (_Float16)y;
            }
        }
    }
}

// Final layer fused with max-over-K; writes q_feats channels directly.
template<int K, int KD, int DOUT>
__device__ inline void run_final_layer(const _Float16* __restrict__ src, const LayerPP L,
                                       float* __restrict__ out, int b, int mq,
                                       int lane, int choff)
{
    const int  m  = lane & 15;
    const bool hi = lane >= 16;
    float mx[DOUT / 16];
#pragma unroll
    for (int t = 0; t < DOUT / 16; ++t) mx[t] = -3.4e38f;

    for (int o0 = 0; o0 < DOUT; o0 += 16) {
        const int t = o0 >> 4;
        const int o = o0 + m;
        const float sc = L.sc[o];
        const float bs = L.bs[o];
        for (int k0 = 0; k0 < K; k0 += 16) {
            v8f acc = {};
#pragma unroll
            for (int i0 = 0; i0 < KD; i0 += 32) {
                v16h a  = load_a_frag(src, KD, k0, i0, lane);
                v16h bf = load_b_frag(L.Wh, KD, o0, i0, lane);
                acc = __builtin_amdgcn_wmma_f32_16x16x32_f16(
                    false, a, false, bf, (short)0, acc, false, false);
            }
#pragma unroll
            for (int j = 0; j < 8; ++j) {
                float y = fmaxf(acc[j] * sc + bs, 0.0f);
                mx[t] = fmaxf(mx[t], y);
            }
        }
    }
#pragma unroll
    for (int t = 0; t < DOUT / 16; ++t) {
        float other = __shfl_xor(mx[t], 16, 32);   // lanes L and L+16 share col
        mx[t] = fmaxf(mx[t], other);
    }
    if (!hi) {
#pragma unroll
        for (int t = 0; t < DOUT / 16; ++t) {
            int ch = choff + t * 16 + m;
            out[(size_t)B_ * 3 * M_ + ((size_t)b * 320 + ch) * M_ + mq] = mx[t];
        }
    }
}

// ---------------------------------------------------------------------------
// Kernel 2 (per scale): fused ordered ball query + grouping + 3-layer MLP
// (f16 WMMA, f32 accum, BN+ReLU folded) + max-pool over K.
// One wave per query, 2 waves per block, per-wave LDS slices, no barriers.
// ---------------------------------------------------------------------------
template<int K, int D1, int D2, int D3, int CHOFF>
__global__ __launch_bounds__(64) void ssg_kernel(const float* __restrict__ pts,
                                                 const float* __restrict__ feats,
                                                 const float* __restrict__ qpts,
                                                 float r2,
                                                 LayerPP L1, LayerPP L2, LayerPP L3,
                                                 float* __restrict__ out)
{
    __shared__ __align__(16) _Float16 sX [2][K * 32];
    __shared__ __align__(16) _Float16 sY1[2][K * D1];
    __shared__ __align__(16) _Float16 sY2[2][K * D2];
    __shared__ int sIdx[2][K];

    const int w    = threadIdx.x >> 5;
    const int lane = threadIdx.x & 31;
    const int q    = blockIdx.x * 2 + w;       // [0, B*M)
    const int b    = q >> 11;                  // q / 2048
    const int mq   = q & 2047;

    const float qx = qpts[q * 3 + 0];
    const float qy = qpts[q * 3 + 1];
    const float qz = qpts[q * 3 + 2];
    const float* px = pts + (size_t)b * 3 * N_;
    const float* py = px + N_;
    const float* pz = py + N_;

    // Ordered ball query: ascending scan + wave32 ballot compaction gives the
    // K lowest-index in-radius points, matching the reference's sort+take-K.
    int cnt = 0;
    for (int base = 0; base < N_ && cnt < K; base += 32) {
        int n = base + lane;
        float dx = px[n] - qx, dy = py[n] - qy, dz = pz[n] - qz;
        bool inr = (dx * dx + dy * dy + dz * dz) <= r2;
        unsigned mask = (unsigned)__ballot(inr);
        int pos = __popc(mask & ((1u << lane) - 1u));
        if (inr) {
            int slot = cnt + pos;
            if (slot < K) sIdx[w][slot] = n;
        }
        cnt += __popc(mask);
    }
    if (cnt > K) cnt = K;
    int first = (cnt > 0) ? sIdx[w][0] : 0;
    for (int k = cnt + lane; k < K; k += 32) sIdx[w][k] = first;

    // Build X tile in LDS: [K][32] f16; cols 0..2 centered coords, 3..18
    // features, 19..31 zero pad (one clean 32-wide WMMA k-chunk).
    for (int t = lane; t < K * 32; t += 32) {
        int k = t >> 5, c = t & 31;
        int n = sIdx[w][k];
        float v;
        if (c < 3)       v = (c == 0) ? (px[n] - qx) : (c == 1) ? (py[n] - qy) : (pz[n] - qz);
        else if (c < 19) v = feats[((size_t)b * C_ + (c - 3)) * N_ + n];
        else             v = 0.0f;
        sX[w][t] = (_Float16)v;
    }

    run_layer<K, 32, D1>(sX[w],  L1, sY1[w], lane);
    run_layer<K, D1, D2>(sY1[w], L2, sY2[w], lane);
    run_final_layer<K, D2, D3>(sY2[w], L3, out, b, mq, lane, CHOFF);
}

// ---------------------------------------------------------------------------
extern "C" void kernel_launch(void* const* d_in, const int* in_sizes, int n_in,
                              void* d_out, int out_size, void* d_ws, size_t ws_size,
                              hipStream_t stream)
{
    (void)in_sizes; (void)n_in; (void)out_size; (void)ws_size;

    const float* s_points = (const float*)d_in[0];
    const float* s_feats  = (const float*)d_in[1];
    float* out = (float*)d_out;

    // ---- workspace carve-up -------------------------------------------------
    char*  base = (char*)d_ws;
    size_t off  = 0;
    float* qpts = (float*)(base + off);               // [B*M][3]
    off += (size_t)B_ * M_ * 3 * sizeof(float);
    off = (off + 255) & ~(size_t)255;

    static const int dinv [9] = {19, 32, 32,  19, 64, 64,  19, 64, 96};
    static const int doutv[9] = {32, 32, 64,  64, 64, 128, 64, 96, 128};

    LayerPP lp[9];
    for (int l = 0; l < 9; ++l) {
        int dp = ((dinv[l] + 31) / 32) * 32;
        _Float16* Wh = (_Float16*)(base + off);
        off += (size_t)doutv[l] * dp * sizeof(_Float16);
        off = (off + 255) & ~(size_t)255;
        float* sc = (float*)(base + off); off += (size_t)doutv[l] * sizeof(float);
        float* bs = (float*)(base + off); off += (size_t)doutv[l] * sizeof(float);
        off = (off + 255) & ~(size_t)255;

        int idx = 2 + l * 5;  // params flattened per layer: W, beta, gamma, mean, var
        const float* W     = (const float*)d_in[idx + 0];
        const float* beta  = (const float*)d_in[idx + 1];
        const float* gamma = (const float*)d_in[idx + 2];
        const float* mean  = (const float*)d_in[idx + 3];
        const float* var   = (const float*)d_in[idx + 4];

        int total = doutv[l] * dp;
        prep_layer<<<(total + 255) / 256, 256, 0, stream>>>(
            W, gamma, beta, mean, var, dinv[l], doutv[l], dp, Wh, sc, bs);

        lp[l].Wh = Wh; lp[l].sc = sc; lp[l].bs = bs;
    }

    // ---- FPS ---------------------------------------------------------------
    fps_kernel<<<B_, 1024, 0, stream>>>(s_points, qpts, out);

    // ---- fused ball-query + MLP + maxpool per scale ------------------------
    const int blocks = (B_ * M_) / 2;
    ssg_kernel<16, 32, 32,  64,   0><<<blocks, 64, 0, stream>>>(
        s_points, s_feats, qpts, 0.01f, lp[0], lp[1], lp[2], out);
    ssg_kernel<32, 64, 64, 128,  64><<<blocks, 64, 0, stream>>>(
        s_points, s_feats, qpts, 0.04f, lp[3], lp[4], lp[5], out);
    ssg_kernel<64, 64, 96, 128, 192><<<blocks, 64, 0, stream>>>(
        s_points, s_feats, qpts, 0.16f, lp[6], lp[7], lp[8], out);
}
